// GNN33_27410481283402
// MI455X (gfx1250) — matrix operand order repaired and checked by reading, not compiled, not run on hardware
//
#include <hip/hip_runtime.h>
#include <stdint.h>

// ---------------------------------------------------------------------------
// Types for CDNA5 WMMA (wave32, 16x16x32 bf16 -> f32)
// ---------------------------------------------------------------------------
typedef __attribute__((ext_vector_type(16))) __bf16 v16bf;
typedef __attribute__((ext_vector_type(16))) short  v16s;
typedef __attribute__((ext_vector_type(8)))  short  v8s;
typedef __attribute__((ext_vector_type(8)))  float  v8f;

#define HEADS   6
#define NNODES  30000
#define NEDGES  480000

static __device__ __forceinline__ uint16_t f2bf(float f) {
    uint32_t u = __builtin_bit_cast(uint32_t, f);
    uint32_t r = u + 0x7FFFu + ((u >> 16) & 1u);   // round-to-nearest-even
    return (uint16_t)(r >> 16);
}

// monotonic float<->uint mapping for atomicMax on floats (handles negatives)
static __device__ __forceinline__ unsigned fkey(float f) {
    unsigned u = __builtin_bit_cast(unsigned, f);
    return (u & 0x80000000u) ? ~u : (u | 0x80000000u);
}
static __device__ __forceinline__ float funkey(unsigned k) {
    unsigned u = (k & 0x80000000u) ? (k ^ 0x80000000u) : ~k;
    return __builtin_bit_cast(float, u);
}

// ---------------------------------------------------------------------------
// Activation padding: h [rows x cin] f32 -> [rows x kpad] bf16 (zero-pad cols)
// ---------------------------------------------------------------------------
__global__ void k_pad_cols(const float* __restrict__ src, uint16_t* __restrict__ dst,
                           int rows, int cin, int kpad) {
    int i = blockIdx.x * blockDim.x + threadIdx.x;
    int total = rows * kpad;
    if (i >= total) return;
    int r = i / kpad, c = i - r * kpad;
    dst[i] = (c < cin) ? f2bf(src[(size_t)r * cin + c]) : (uint16_t)0;
}

// ---------------------------------------------------------------------------
// Weight pack: W [kin x N] f32 -> WMMA B-fragment layout, bf16, K zero-padded.
// Element ((nt*ktiles + kt)*32 + lane)*16 + e  <-  B[kt*32 + (lane>>4)*8 + e + (e>>3)*8][nt*16 + (lane&15)]
// so each lane's 16 B elements for one (kt,nt) fragment are contiguous (32 B).
// ---------------------------------------------------------------------------
__global__ void k_pack_w(const float* __restrict__ W, uint16_t* __restrict__ Bp,
                         int kin, int K, int N) {
    int i = blockIdx.x * blockDim.x + threadIdx.x;
    int total = K * N;
    if (i >= total) return;
    int e    = i & 15;
    int lane = (i >> 4) & 31;
    int t    = i >> 9;                 // nt*ktiles + kt
    int ktiles = K >> 5;
    int nt = t / ktiles, kt = t - nt * ktiles;
    int l = lane & 15, hi = lane >> 4;
    int k = (kt << 5) + (hi << 3) + e + ((e >> 3) << 3);
    int n = (nt << 4) + l;
    Bp[i] = (k < kin) ? f2bf(W[(size_t)k * N + n]) : (uint16_t)0;
}

// ---------------------------------------------------------------------------
// GEMM: C[MxN] = A[MxK](bf16 row-major) * B(packed fragments), fp32 accum.
// One wave32 per 16x16 tile. Inner loop: 2x b128 (A) + 2x b128 (B) + 1 WMMA.
// ---------------------------------------------------------------------------
__global__ void k_gemm_wmma(const uint16_t* __restrict__ A,
                            const uint16_t* __restrict__ Bp,
                            float* __restrict__ C,
                            int M, int K, int N) {
    int wave = (int)((blockIdx.x * (size_t)blockDim.x + threadIdx.x) >> 5);
    int lane = threadIdx.x & 31;
    int tilesN = N >> 4;
    int tiles  = (M >> 4) * tilesN;
    if (wave >= tiles) return;               // uniform per wave -> EXEC all-1s at WMMA
    int tm = wave / tilesN, tn = wave - tm * tilesN;
    int m0 = tm << 4, n0 = tn << 4;
    int l  = lane & 15, hi = lane >> 4;
    int ktiles = K >> 5;

    // A fragment rows (ISA 7.12.2, 16-bit A 16x32): lane row = m0+l,
    // lanes 0-15: K = 0..7 & 16..23 ; lanes 16-31: K = 8..15 & 24..31
    const uint16_t* arow  = A  + (size_t)(m0 + l) * K + hi * 8;
    const uint16_t* bfrag = Bp + ((size_t)tn * ktiles * 32 + lane) * 16;

    v8f acc = {};
    for (int kt = 0; kt < ktiles; ++kt) {
        v8s alo = *(const v8s*)(arow);         // K = hi*8 + 0..7
        v8s ahi = *(const v8s*)(arow + 16);    // K = 16 + hi*8 + 0..7
        v16s av = __builtin_shufflevector(alo, ahi,
                 0, 1, 2, 3, 4, 5, 6, 7, 8, 9, 10, 11, 12, 13, 14, 15);
        v16s bv = *(const v16s*)(bfrag);       // pre-packed: contiguous 32 B
        acc = __builtin_amdgcn_wmma_f32_16x16x32_bf16(
            false, __builtin_bit_cast(v16bf, av),
            false, __builtin_bit_cast(v16bf, bv),
            (short)0, acc, false, false);
        arow  += 32;
        bfrag += 512;                           // 32 lanes * 16 elems
    }
    // C/D layout: VGPR r -> row m0 + r + 8*hi, col n0 + l
#pragma unroll
    for (int r = 0; r < 8; ++r)
        C[(size_t)(m0 + r + 8 * hi) * N + n0 + l] = acc[r];
}

// ---------------------------------------------------------------------------
// Attention scores per node/head: e_src/e_dst = <z[n,h,:], a_{s,d}[h,:]>
// ---------------------------------------------------------------------------
__global__ void k_scores(const float* __restrict__ z, const float* __restrict__ as_,
                         const float* __restrict__ ad_, float* __restrict__ esrc,
                         float* __restrict__ edst, int N, int Fo) {
    int i = blockIdx.x * blockDim.x + threadIdx.x;
    if (i >= N * HEADS) return;
    int n = i / HEADS, h = i - n * HEADS;
    const float* zp = z + ((size_t)n * HEADS + h) * Fo;
    float s = 0.f, d = 0.f;
    for (int f = 0; f < Fo; ++f) {
        s += zp[f] * as_[h * Fo + f];
        d += zp[f] * ad_[h * Fo + f];
    }
    esrc[i] = s; edst[i] = d;
}

// ---------------------------------------------------------------------------
// Edge passes: segment-max, exp + segment-sum, normalize, aggregate
// ---------------------------------------------------------------------------
static __device__ __forceinline__ float edge_logit(const float* esrc, const float* edst,
                                                   int s, int d, int h) {
    float v = esrc[s * HEADS + h] + edst[d * HEADS + h];
    return v > 0.f ? v : 0.2f * v;            // leaky_relu(0.2)
}

__global__ void k_edge_max(const int* __restrict__ src, const int* __restrict__ dst,
                           const float* __restrict__ esrc, const float* __restrict__ edst,
                           unsigned* __restrict__ mmax, int E) {
    int i = blockIdx.x * blockDim.x + threadIdx.x;
    if (i >= E * HEADS) return;
    int e = i / HEADS, h = i - e * HEADS;
    int s = src[e], d = dst[e];
    atomicMax(&mmax[d * HEADS + h], fkey(edge_logit(esrc, edst, s, d, h)));
}

__global__ void k_edge_exp(const int* __restrict__ src, const int* __restrict__ dst,
                           const float* __restrict__ esrc, const float* __restrict__ edst,
                           const unsigned* __restrict__ mmax, float* __restrict__ exbuf,
                           float* __restrict__ denom, int E) {
    int i = blockIdx.x * blockDim.x + threadIdx.x;
    if (i >= E * HEADS) return;
    int e = i / HEADS, h = i - e * HEADS;
    int s = src[e], d = dst[e];
    float v  = edge_logit(esrc, edst, s, d, h);
    float ex = expf(v - funkey(mmax[d * HEADS + h]));
    exbuf[i] = ex;
    atomicAdd(&denom[d * HEADS + h], ex);
}

__global__ void k_edge_alpha(const int* __restrict__ dst, float* __restrict__ exbuf,
                             const float* __restrict__ denom, int E) {
    int i = blockIdx.x * blockDim.x + threadIdx.x;
    if (i >= E * HEADS) return;
    int e = i / HEADS, h = i - e * HEADS;
    exbuf[i] = exbuf[i] / (denom[dst[e] * HEADS + h] + 1e-16f);
}

// out[dst, f..f+3] += z[src, f..f+3] * alpha[e, head(f)]
__global__ void k_agg(const int* __restrict__ src, const int* __restrict__ dst,
                      const float* __restrict__ z, const float* __restrict__ alpha,
                      float* __restrict__ out, int E, int F, int Fo) {
    int i = blockIdx.x * blockDim.x + threadIdx.x;
    int chunks = F >> 2;
    if (i >= E * chunks) return;
    int e = i / chunks, q = i - e * chunks;
    int f = q << 2;
    int h = f / Fo;                            // Fo multiple of 4 -> uniform head
    float a = alpha[e * HEADS + h];
    int s = src[e], d = dst[e];
    const float* zp = z + (size_t)s * F + f;
    __builtin_prefetch(zp, 0, 3);
    float4 zv = *(const float4*)zp;
    float* op = out + (size_t)d * F + f;
    atomicAdd(op + 0, zv.x * a);
    atomicAdd(op + 1, zv.y * a);
    atomicAdd(op + 2, zv.z * a);
    atomicAdd(op + 3, zv.w * a);
}

// ELU in place (fp32) + bf16 copy (padded) for next-layer GEMM
__global__ void k_elu_cvt(float* __restrict__ h, uint16_t* __restrict__ hb,
                          int N, int F, int kpad) {
    int i = blockIdx.x * blockDim.x + threadIdx.x;
    int total = N * kpad;
    if (i >= total) return;
    int n = i / kpad, c = i - n * kpad;
    if (c < F) {
        float x = h[(size_t)n * F + c];
        x = x > 0.f ? x : expf(x) - 1.f;
        h[(size_t)n * F + c] = x;
        hb[i] = f2bf(x);
    } else {
        hb[i] = 0;
    }
}

// column-wise sum pooling: one block per column
__global__ void k_pool(const float* __restrict__ h, float* __restrict__ pooled,
                       int N, int F) {
    int c = blockIdx.x;
    float s = 0.f;
    for (int n = threadIdx.x; n < N; n += blockDim.x) s += h[(size_t)n * F + c];
    __shared__ float sm[256];
    sm[threadIdx.x] = s; __syncthreads();
    for (int st = 128; st > 0; st >>= 1) {
        if (threadIdx.x < st) sm[threadIdx.x] += sm[threadIdx.x + st];
        __syncthreads();
    }
    if (threadIdx.x == 0) pooled[c] = sm[0];
}

// final: normalize pooled[768] and dot with Wd, add bd
__global__ void k_final(const float* __restrict__ pooled, const float* __restrict__ Wd,
                        const float* __restrict__ bd, float* __restrict__ out) {
    __shared__ float s_n[256], s_d[256];
    float nrm = 0.f, dot = 0.f;
    for (int i = threadIdx.x; i < 768; i += 256) {
        float v = pooled[i];
        nrm += v * v;
        dot += v * Wd[i];
    }
    s_n[threadIdx.x] = nrm; s_d[threadIdx.x] = dot; __syncthreads();
    for (int st = 128; st > 0; st >>= 1) {
        if (threadIdx.x < st) {
            s_n[threadIdx.x] += s_n[threadIdx.x + st];
            s_d[threadIdx.x] += s_d[threadIdx.x + st];
        }
        __syncthreads();
    }
    if (threadIdx.x == 0) {
        float nm = fmaxf(sqrtf(s_n[0]), 1e-12f);
        out[0] = s_d[0] / nm + bd[0];
    }
}

// ---------------------------------------------------------------------------
// Host launch
// ---------------------------------------------------------------------------
static inline unsigned div_up(size_t a, size_t b) { return (unsigned)((a + b - 1) / b); }

extern "C" void kernel_launch(void* const* d_in, const int* in_sizes, int n_in,
                              void* d_out, int out_size, void* d_ws, size_t ws_size,
                              hipStream_t stream) {
    (void)in_sizes; (void)n_in; (void)out_size; (void)ws_size;
    const int N = NNODES, E = NEDGES;
    const int FO[3]   = {16, 32, 64};
    const int FOUT[3] = {96, 192, 384};
    const int KPAD[3] = {32, 96, 192};
    const int KIN[3]  = {11, 96, 192};

    const float* x_g[2]  = {(const float*)d_in[0], (const float*)d_in[1]};
    const int*   ei_g[2] = {(const int*)d_in[2], (const int*)d_in[3]};
    const float* W[3]    = {(const float*)d_in[4], (const float*)d_in[7], (const float*)d_in[10]};
    const float* As[3]   = {(const float*)d_in[5], (const float*)d_in[8], (const float*)d_in[11]};
    const float* Ad[3]   = {(const float*)d_in[6], (const float*)d_in[9], (const float*)d_in[12]};
    const float* Wd      = (const float*)d_in[13];
    const float* bd      = (const float*)d_in[14];

    // workspace layout (256B aligned)
    char* p = (char*)d_ws;
    auto take = [&](size_t bytes) { char* r = p; p += (bytes + 255) & ~(size_t)255; return r; };
    uint16_t* hb    = (uint16_t*)take((size_t)N * 384 * 2);   // bf16 activations (padded)
    float*    z     = (float*)take((size_t)N * 384 * 4);      // GEMM output
    float*    outb  = (float*)take((size_t)N * 384 * 4);      // aggregated / ELU'd h
    float*    esrc  = (float*)take((size_t)N * HEADS * 4);
    float*    edst  = (float*)take((size_t)N * HEADS * 4);
    unsigned* mmax  = (unsigned*)take((size_t)N * HEADS * 4);
    float*    denom = (float*)take((size_t)N * HEADS * 4);
    float*    exbuf = (float*)take((size_t)E * HEADS * 4);
    uint16_t* Wb[3];
    for (int l = 0; l < 3; ++l) Wb[l] = (uint16_t*)take((size_t)KPAD[l] * FOUT[l] * 2);
    float*    pooled = (float*)take(768 * 4);

    const int T = 256;

    // pack weights into WMMA B-fragment layout (bf16, K zero-padded)
    for (int l = 0; l < 3; ++l) {
        int total = KPAD[l] * FOUT[l];
        k_pack_w<<<div_up(total, T), T, 0, stream>>>(W[l], Wb[l], KIN[l], KPAD[l], FOUT[l]);
    }
    hipMemsetAsync(pooled, 0, 768 * 4, stream);

    for (int g = 0; g < 2; ++g) {
        const float* x  = x_g[g];
        const int* srcp = ei_g[g];
        const int* dstp = ei_g[g] + E;

        // layer-1 input: pad 11 -> 32 columns, fp32 -> bf16
        k_pad_cols<<<div_up((size_t)N * 32, T), T, 0, stream>>>(x, hb, N, 11, 32);

        for (int l = 0; l < 3; ++l) {
            const int Fo = FO[l], F = FOUT[l], K = KPAD[l];

            // GEMM (WMMA bf16): z = h @ W
            {
                size_t tiles = (size_t)(N / 16) * (F / 16);
                k_gemm_wmma<<<div_up(tiles * 32, T), T, 0, stream>>>(hb, Wb[l], z, N, K, F);
            }
            // per-node attention scores
            k_scores<<<div_up((size_t)N * HEADS, T), T, 0, stream>>>(z, As[l], Ad[l], esrc, edst, N, Fo);

            // reset segment accumulators
            hipMemsetAsync(mmax,  0, (size_t)N * HEADS * 4, stream);   // key(x) >= 0 for all finite x
            hipMemsetAsync(denom, 0, (size_t)N * HEADS * 4, stream);
            hipMemsetAsync(outb,  0, (size_t)N * F * 4, stream);

            unsigned ehb = div_up((size_t)E * HEADS, T);
            k_edge_max  <<<ehb, T, 0, stream>>>(srcp, dstp, esrc, edst, mmax, E);
            k_edge_exp  <<<ehb, T, 0, stream>>>(srcp, dstp, esrc, edst, mmax, exbuf, denom, E);
            k_edge_alpha<<<ehb, T, 0, stream>>>(dstp, exbuf, denom, E);
            k_agg<<<div_up((size_t)E * (F / 4), T), T, 0, stream>>>(srcp, dstp, z, exbuf, outb, E, F, Fo);

            // ELU + bf16 for next layer (next K = F, already multiple of 32)
            k_elu_cvt<<<div_up((size_t)N * F, T), T, 0, stream>>>(outb, hb, N, F, F);
        }
        // sum pooling of final fp32 h (in outb, F=384)
        k_pool<<<384, 256, 0, stream>>>(outb, pooled + g * 384, N, 384);
    }
    k_final<<<1, 256, 0, stream>>>(pooled, Wd, bd, (float*)d_out);
}
